// GATLSTMPointerNetwork_35991825940628
// MI455X (gfx1250) — matrix operand order repaired and checked by reading, not compiled, not run on hardware
//
#include <hip/hip_runtime.h>

// ---------------- problem dims ----------------
constexpr int NB = 64, SEQ = 200, HD = 256, NHEAD = 8, GL = 2, LAYERS = 3,
              FFD = 512, HDH = 32;

typedef __attribute__((ext_vector_type(16))) _Float16 v16h;
typedef __attribute__((ext_vector_type(8)))  float    v8f;

// ---------------- generic batched strided WMMA GEMM ----------------
// C(z,m,n) = act( alpha * sum_k A(z,m,k)*B(z,k,n) + bias[n] + Cin(z,m,n) )
// A(z,m,k) = A[z*sA + m*lda + k]
// B(z,k,n) = Bm[z*sB + k*ldbK + n*ldbN]          (strided -> free transpose)
// C(z,m,n) = C[(z/cdiv)*sC1 + (z%cdiv)*sC2 + m*ldc + n]   (or QKV head-split)
struct GemmP {
  const float* A; const float* Bm; const float* bias; const float* Cin; float* C;
  int M, N, K;
  int lda; long long sA;
  int ldbK, ldbN; long long sB;
  int ldc; long long sC1, sC2; int cdiv;
  float alpha; int flags; // 1=bias 2=relu 4=residual 8=qkv head-split output
};

#define TM 128
#define TN 128
#define TK 32
#define AP 40   // LDS row pitch (halves): 16B-aligned fragments, bank-friendly
#define BP 40

union H8 { _Float16 h[8]; int4 q; };

__global__ __launch_bounds__(256) void gemm_wmma(GemmP p) {
  __shared__ _Float16 As[TM * AP];
  __shared__ _Float16 Bs[TN * BP];
  const int tid = threadIdx.x;
  const int lane = tid & 31, wid = tid >> 5;
  const int wm = (wid & 3) * 32, wn = (wid >> 2) * 64;   // 4x2 wave grid, 32x64/wave
  const int bm = blockIdx.y * TM, bn = blockIdx.x * TN;
  const int z = blockIdx.z;
  const long long offA = (long long)z * p.sA;
  const long long offB = (long long)z * p.sB;
  const long long offC =
      (long long)(z / p.cdiv) * p.sC1 + (long long)(z % p.cdiv) * p.sC2;

  v8f acc[2][4] = {};

  const int ar = tid >> 1, akb = (tid & 1) * 16;   // A staging: 16 elems/thread
  const int l16 = lane & 15, hi = lane >> 4;

  for (int kt = 0; kt < p.K; kt += TK) {
    { // ---- stage A tile (f32 -> f16, WMMA A-fragment k-permuted layout) ----
      const int gm = bm + ar;
      const int c0 = akb >> 1, c1 = (akb >> 1) + 16;  // permuted 8-chunk bases
      _Float16* dst = As + ar * AP;
      if (gm < p.M && (kt + TK) <= p.K) {
        const float4* s4 =
            (const float4*)(p.A + offA + (long long)gm * p.lda + kt + akb);
        const float4 f0 = s4[0], f1 = s4[1], f2 = s4[2], f3 = s4[3];
        H8 h0, h1;
        h0.h[0] = (_Float16)f0.x; h0.h[1] = (_Float16)f0.y;
        h0.h[2] = (_Float16)f0.z; h0.h[3] = (_Float16)f0.w;
        h0.h[4] = (_Float16)f1.x; h0.h[5] = (_Float16)f1.y;
        h0.h[6] = (_Float16)f1.z; h0.h[7] = (_Float16)f1.w;
        h1.h[0] = (_Float16)f2.x; h1.h[1] = (_Float16)f2.y;
        h1.h[2] = (_Float16)f2.z; h1.h[3] = (_Float16)f2.w;
        h1.h[4] = (_Float16)f3.x; h1.h[5] = (_Float16)f3.y;
        h1.h[6] = (_Float16)f3.z; h1.h[7] = (_Float16)f3.w;
        *(int4*)(dst + c0) = h0.q;
        *(int4*)(dst + c1) = h1.q;
        if ((kt + 2 * TK) <= p.K) __builtin_prefetch(s4 + 8, 0, 0);
      } else { // M / K edge: guarded scalar
        const float* src = p.A + offA + (long long)gm * p.lda + kt + akb;
        #pragma unroll
        for (int j = 0; j < 16; ++j) {
          const int k = akb + j;
          float v = (gm < p.M && (kt + k) < p.K) ? src[j] : 0.0f;
          const int pp = (k >= 8 && k < 24) ? (k ^ 24) : k;
          dst[pp] = (_Float16)v;
        }
      }
    }
    { // ---- stage B tile n-major: Bs[n][k] ----
      if (p.ldbK == 1) {          // memory contiguous along k
        const int nn = tid & 127, kb = (tid >> 7) * 16;
        const int gn = bn + nn;
        _Float16* dst = Bs + nn * BP + kb;
        if (gn < p.N && (kt + TK) <= p.K) {
          const float4* s4 =
              (const float4*)(p.Bm + offB + kt + kb + (long long)gn * p.ldbN);
          const float4 f0 = s4[0], f1 = s4[1], f2 = s4[2], f3 = s4[3];
          H8 h0, h1;
          h0.h[0] = (_Float16)f0.x; h0.h[1] = (_Float16)f0.y;
          h0.h[2] = (_Float16)f0.z; h0.h[3] = (_Float16)f0.w;
          h0.h[4] = (_Float16)f1.x; h0.h[5] = (_Float16)f1.y;
          h0.h[6] = (_Float16)f1.z; h0.h[7] = (_Float16)f1.w;
          h1.h[0] = (_Float16)f2.x; h1.h[1] = (_Float16)f2.y;
          h1.h[2] = (_Float16)f2.z; h1.h[3] = (_Float16)f2.w;
          h1.h[4] = (_Float16)f3.x; h1.h[5] = (_Float16)f3.y;
          h1.h[6] = (_Float16)f3.z; h1.h[7] = (_Float16)f3.w;
          *(int4*)(dst) = h0.q;
          *(int4*)(dst + 8) = h1.q;
        } else {
          #pragma unroll
          for (int j = 0; j < 16; ++j) {
            const int k = kb + j;
            float v = (gn < p.N && (kt + k) < p.K)
                          ? p.Bm[offB + (kt + k) + (long long)gn * p.ldbN]
                          : 0.0f;
            dst[j] = (_Float16)v;
          }
        }
      } else if (p.ldbN == 1) {   // memory contiguous along n
        const int kk = tid & 31, nb = (tid >> 5) * 16;
        if ((bn + nb + 16) <= p.N && (kt + kk) < p.K) {
          const float4* s4 = (const float4*)(p.Bm + offB +
              (long long)(kt + kk) * p.ldbK + bn + nb);
          const float4 f0 = s4[0], f1 = s4[1], f2 = s4[2], f3 = s4[3];
          _Float16 hh[16];
          hh[0]  = (_Float16)f0.x; hh[1]  = (_Float16)f0.y;
          hh[2]  = (_Float16)f0.z; hh[3]  = (_Float16)f0.w;
          hh[4]  = (_Float16)f1.x; hh[5]  = (_Float16)f1.y;
          hh[6]  = (_Float16)f1.z; hh[7]  = (_Float16)f1.w;
          hh[8]  = (_Float16)f2.x; hh[9]  = (_Float16)f2.y;
          hh[10] = (_Float16)f2.z; hh[11] = (_Float16)f2.w;
          hh[12] = (_Float16)f3.x; hh[13] = (_Float16)f3.y;
          hh[14] = (_Float16)f3.z; hh[15] = (_Float16)f3.w;
          #pragma unroll
          for (int j = 0; j < 16; ++j) Bs[(nb + j) * BP + kk] = hh[j];
        } else {
          #pragma unroll
          for (int j = 0; j < 16; ++j) {
            const int gn = bn + nb + j;
            float v = (gn < p.N && (kt + kk) < p.K)
                          ? p.Bm[offB + (long long)(kt + kk) * p.ldbK + gn]
                          : 0.0f;
            Bs[(nb + j) * BP + kk] = (_Float16)v;
          }
        }
      } else {                    // generic strided scalar
        const int nn = tid & 127, kb = (tid >> 7) * 16;
        const int gn = bn + nn;
        _Float16* dst = Bs + nn * BP + kb;
        #pragma unroll
        for (int j = 0; j < 16; ++j) {
          const int k = kb + j;
          float v = (gn < p.N && (kt + k) < p.K)
              ? p.Bm[offB + (long long)(kt + k) * p.ldbK + (long long)gn * p.ldbN]
              : 0.0f;
          dst[j] = (_Float16)v;
        }
      }
    }
    __syncthreads();
    { // ---- compute: 2 A-frags x 4 B-frags -> 8 WMMAs ----
      union FragU { int4 q[2]; v16h v; };
      FragU fa[2], fb[4];
      #pragma unroll
      for (int mi = 0; mi < 2; ++mi) {
        const int4* pa = (const int4*)(As + (wm + mi * 16 + l16) * AP + hi * 16);
        fa[mi].q[0] = pa[0]; fa[mi].q[1] = pa[1];
      }
      #pragma unroll
      for (int ni = 0; ni < 4; ++ni) {
        const int4* pb = (const int4*)(Bs + (wn + ni * 16 + l16) * BP + hi * 8);
        fb[ni].q[0] = pb[0]; fb[ni].q[1] = pb[2];
      }
      #pragma unroll
      for (int mi = 0; mi < 2; ++mi)
        #pragma unroll
        for (int ni = 0; ni < 4; ++ni)
          acc[mi][ni] = __builtin_amdgcn_wmma_f32_16x16x32_f16(
              false, fa[mi].v, false, fb[ni].v, (short)0, acc[mi][ni],
              false, false);
    }
    __syncthreads();
  }

  // epilogue: C layout VGPR r -> (lanes0-15 M=r, lanes16-31 M=8+r), N=lane%16
  #pragma unroll
  for (int mi = 0; mi < 2; ++mi)
    #pragma unroll
    for (int ni = 0; ni < 4; ++ni)
      #pragma unroll
      for (int r = 0; r < 8; ++r) {
        const int row = bm + wm + mi * 16 + hi * 8 + r;
        const int col = bn + wn + ni * 16 + l16;
        if (row < p.M && col < p.N) {
          float v = acc[mi][ni][r] * p.alpha;
          if (p.flags & 1) v += p.bias[col];
          long long idx;
          if (p.flags & 8) { // write [B,NH,SEQ,Dh] directly
            const int b = row / SEQ, qn = row % SEQ, h = col >> 5, dh = col & 31;
            idx = (((long long)(b * NHEAD + h) * SEQ + qn) * HDH) + dh;
          } else {
            idx = offC + (long long)row * p.ldc + col;
          }
          if (p.flags & 4) v += p.Cin[idx];
          if (p.flags & 2) v = v > 0.0f ? v : 0.0f;
          p.C[idx] = v;
        }
      }
}

// ---------------- wave32 reductions ----------------
__device__ inline float waveSum(float v) {
  #pragma unroll
  for (int o = 16; o; o >>= 1) v += __shfl_xor(v, o, 32);
  return v;
}
__device__ inline float waveMax(float v) {
  #pragma unroll
  for (int o = 16; o; o >>= 1) v = fmaxf(v, __shfl_xor(v, o, 32));
  return v;
}
__device__ inline float blockRed(float v, float* red, bool ismax) {
  const int lane = threadIdx.x & 31, w = threadIdx.x >> 5;
  v = ismax ? waveMax(v) : waveSum(v);
  __syncthreads();
  if (lane == 0) red[w] = v;
  __syncthreads();
  float t = red[0];
  #pragma unroll
  for (int q = 1; q < 8; ++q) t = ismax ? fmaxf(t, red[q]) : (t + red[q]);
  return t;
}

// ---------------- GAT attention coefficients ----------------
__global__ __launch_bounds__(256)
void gat_dots(const float* xp, const float* wsrc, const float* wdst,
              float* asrc, float* adst) {
  const int w = blockIdx.x * 8 + (threadIdx.x >> 5); // (b*SEQ+n)*NHEAD+h
  if (w >= NB * SEQ * NHEAD) return;
  const int lane = threadIdx.x & 31;
  const int h = w % NHEAD;
  const float* x = xp + (long long)w * HD;
  const float* ws = wsrc + h * HD;
  const float* wd = wdst + h * HD;
  float s = 0.f, d = 0.f;
  for (int j = lane; j < HD; j += 32) { float v = x[j]; s += v * ws[j]; d += v * wd[j]; }
  s = waveSum(s); d = waveSum(d);
  if (lane == 0) { asrc[w] = s; adst[w] = d; }
}

__global__ __launch_bounds__(256)
void gat_softmax(const float* asrc, const float* adst, const int* adj, float* attn) {
  const int bhj = blockIdx.x;                   // (b*NHEAD+h)*SEQ + j
  const int j = bhj % SEQ, bh = bhj / SEQ;
  const int h = bh % NHEAD, b = bh / NHEAD;
  const int i = threadIdx.x;
  const bool act = i < SEQ;
  float lg = -1e30f;
  if (act) {
    const bool m = (i == j) || (adj[((long long)b * SEQ + i) * SEQ + j] > 0);
    if (m) {
      float t = adst[((long long)b * SEQ + j) * NHEAD + h] +
                asrc[((long long)b * SEQ + i) * NHEAD + h];
      lg = t > 0.f ? t : 0.2f * t;              // leaky_relu(0.2)
    } else lg = -1e9f;
  }
  __shared__ float red[8];
  const float mx = blockRed(lg, red, true);
  const float e = act ? __expf(lg - mx) : 0.f;
  const float sm = blockRed(e, red, false);
  if (act) attn[(long long)bhj * SEQ + i] = e / sm;
}

__global__ __launch_bounds__(256)
void gat_combine(float* x, const float* msg, const float* bias, long long n) {
  const long long i = (long long)blockIdx.x * 256 + threadIdx.x;
  if (i < n) {
    float v = msg[i] * (1.0f / NHEAD) + bias[i % HD];
    v = v > 0.f ? v : 0.f;
    x[i] += v;
  }
}

// ---------------- LayerNorm (one 256-thread block per 256-wide row) ----------------
__global__ __launch_bounds__(256)
void layernorm(const float* a, const float* bsrc, const float* gamma,
               const float* beta, float* out, int relu_b) {
  const long long row = blockIdx.x;
  const int c = threadIdx.x;
  float v = a[row * HD + c];
  if (bsrc) { float w = bsrc[row * HD + c]; if (relu_b) w = w > 0.f ? w : 0.f; v += w; }
  __shared__ float red[8];
  const float mean = blockRed(v, red, false) * (1.f / HD);
  const float d = v - mean;
  const float var = blockRed(d * d, red, false) * (1.f / HD);
  out[row * HD + c] = d * __frsqrt_rn(var + 1e-5f) * gamma[c] + beta[c];
}

// ---------------- masked attention softmax (in place) ----------------
__global__ __launch_bounds__(256)
void attn_softmax(float* sc, const int* adj) {
  const int bhq = blockIdx.x;                  // (b*NHEAD+h)*SEQ + q
  const int q = bhq % SEQ, b = bhq / (NHEAD * SEQ);
  const int k = threadIdx.x;
  const bool act = k < SEQ;
  const long long idx = (long long)bhq * SEQ + k;
  float lg = -1e30f;
  if (act) {
    const bool m = adj[((long long)b * SEQ + q) * SEQ + k] > 0;
    lg = m ? sc[idx] : -1e9f;
  }
  __shared__ float red[8];
  const float mx = blockRed(lg, red, true);
  const float e = act ? __expf(lg - mx) : 0.f;
  const float s = blockRed(e, red, false);
  if (act) sc[idx] = e / s;
}

// ---------------- LSTM cell pointwise ----------------
__global__ __launch_bounds__(256)
void lstm_elem(const float* gates, const float* dec_c, float* hout, float* cout) {
  const int i = blockIdx.x * 256 + threadIdx.x;
  if (i >= NB * HD) return;
  const int b = i / HD, n = i % HD;
  const float* g = gates + (long long)b * 4 * HD;
  const float ig = g[n], fg = g[HD + n], gg = g[2 * HD + n], og = g[3 * HD + n];
  const float si = 1.f / (1.f + __expf(-ig));
  const float sf = 1.f / (1.f + __expf(-fg));
  const float so = 1.f / (1.f + __expf(-og));
  const float c = sf * dec_c[i] + si * tanhf(gg);
  cout[i] = c;
  hout[i] = so * tanhf(c);
}

// ---------------- pointer logits + policy ----------------
__global__ __launch_bounds__(256)
void ptr_u(const float* ref, const float* qh, const float* v, const int* mask,
           float* u) {
  const int w = blockIdx.x * 8 + (threadIdx.x >> 5); // b*SEQ + n
  if (w >= NB * SEQ) return;
  const int lane = threadIdx.x & 31;
  const int b = w / SEQ;
  const float* r = ref + (long long)w * HD;
  const float* q = qh + (long long)b * HD;
  float s = 0.f;
  for (int j = lane; j < HD; j += 32) s += tanhf(r[j] + q[j]) * v[j];
  s = waveSum(s);
  if (lane == 0) u[w] = (mask[w] > 0) ? s : -1e9f;
}

__global__ __launch_bounds__(256)
void policy_softmax(const float* u, float* pol) {
  const int b = blockIdx.x, n = threadIdx.x;
  const bool act = n < SEQ;
  float lg = act ? u[b * SEQ + n] : -1e30f;
  __shared__ float red[8];
  const float mx = blockRed(lg, red, true);
  const float e = act ? __expf(lg - mx) : 0.f;
  const float s = blockRed(e, red, false);
  if (act) pol[b * SEQ + n] = e / s;
}

// ---------------- host orchestration ----------------
static inline void launch_gemm(hipStream_t st, const float* A, const float* Bm,
    const float* bias, const float* Cin, float* C, int M, int N, int K,
    int lda, long long sA, int ldbK, int ldbN, long long sB,
    int ldc, long long sC1, long long sC2, int cdiv, int batch,
    float alpha, int flags) {
  GemmP p;
  p.A = A; p.Bm = Bm; p.bias = bias; p.Cin = Cin; p.C = C;
  p.M = M; p.N = N; p.K = K;
  p.lda = lda; p.sA = sA; p.ldbK = ldbK; p.ldbN = ldbN; p.sB = sB;
  p.ldc = ldc; p.sC1 = sC1; p.sC2 = sC2; p.cdiv = cdiv;
  p.alpha = alpha; p.flags = flags;
  dim3 grid((N + TN - 1) / TN, (M + TM - 1) / TM, batch);
  gemm_wmma<<<grid, 256, 0, st>>>(p);
}

extern "C" void kernel_launch(void* const* d_in, const int* in_sizes, int n_in,
                              void* d_out, int out_size, void* d_ws, size_t ws_size,
                              hipStream_t stream) {
  (void)in_sizes; (void)n_in; (void)out_size; (void)ws_size;
  const float* enc      = (const float*)d_in[0];
  const float* dec_in   = (const float*)d_in[1];
  const float* dec_h    = (const float*)d_in[2];
  const float* dec_c    = (const float*)d_in[3];
  const float* gat_W    = (const float*)d_in[4];
  const float* gat_as   = (const float*)d_in[5];
  const float* gat_ad   = (const float*)d_in[6];
  const float* gat_b    = (const float*)d_in[7];
  const float* gat_ln   = (const float*)d_in[8];
  const float* attn_W   = (const float*)d_in[9];
  const float* attn_b   = (const float*)d_in[10];
  const float* ln1      = (const float*)d_in[11];
  const float* ffn_W1   = (const float*)d_in[12];
  const float* ffn_b1   = (const float*)d_in[13];
  const float* ffn_W2   = (const float*)d_in[14];
  const float* ffn_b2   = (const float*)d_in[15];
  const float* ln2      = (const float*)d_in[16];
  const float* lstm_Wih = (const float*)d_in[17];
  const float* lstm_Whh = (const float*)d_in[18];
  const float* lstm_b   = (const float*)d_in[19];
  const float* ptr_Wref = (const float*)d_in[20];
  const float* ptr_Wq   = (const float*)d_in[21];
  const float* ptr_v    = (const float*)d_in[22];
  const int*   adj      = (const int*)d_in[23];
  const int*   maskp    = (const int*)d_in[24];

  // workspace layout (floats), ~204 MB total
  float* ws   = (float*)d_ws;
  const size_t XPn  = (size_t)NB * SEQ * NHEAD * HD;  // 26,214,400
  const size_t ATTn = (size_t)NB * NHEAD * SEQ * SEQ; // 20,480,000
  const size_t Xn   = (size_t)NB * SEQ * HD;          //  3,276,800
  float* XP   = ws;
  float* ATT  = XP + XPn;
  float* X    = ATT + ATTn;
  float* MSG  = X + Xn;
  float* ASRC = MSG + Xn;
  float* ADST = ASRC + (size_t)NB * SEQ * NHEAD;
  float* GATES= ADST + (size_t)NB * SEQ * NHEAD;
  float* QH   = GATES + (size_t)NB * 4 * HD;
  float* UU   = QH + (size_t)NB * HD;
  // transformer scratch reuses XP
  float* Q   = XP;
  float* Kt  = XP + Xn;
  float* V   = XP + 2 * Xn;
  float* CTX = XP + 3 * Xn;
  float* FF  = XP + 4 * Xn;

  float* out = (float*)d_out;
  float* POL = out;                          // [B,SEQ]
  float* HO  = out + (size_t)NB * SEQ;       // [B,H]
  float* CO  = HO + (size_t)NB * HD;         // [B,H]
  float* EMB = CO + (size_t)NB * HD;         // [B,SEQ,H]

  // ---- GAT stack ----
  hipMemcpyAsync(X, enc, sizeof(float) * Xn, hipMemcpyDeviceToDevice, stream);
  for (int g = 0; g < GL; ++g) {
    // xp = x @ W[g]  : [12800,2048] = [12800,256]@[256,2048]
    launch_gemm(stream, X, gat_W + (size_t)g * HD * NHEAD * HD, nullptr, nullptr,
                XP, NB * SEQ, NHEAD * HD, HD,
                HD, 0, NHEAD * HD, 1, 0,
                NHEAD * HD, 0, 0, 1, 1, 1.0f, 0);
    gat_dots<<<(NB * SEQ * NHEAD + 7) / 8, 256, 0, stream>>>(
        XP, gat_as + (size_t)g * NHEAD * HD, gat_ad + (size_t)g * NHEAD * HD,
        ASRC, ADST);
    gat_softmax<<<NB * NHEAD * SEQ, 256, 0, stream>>>(ASRC, ADST, adj, ATT);
    // msg mean: accumulate per-head batched GEMMs into MSG [B,SEQ,HD]
    for (int h = 0; h < NHEAD; ++h) {
      launch_gemm(stream, ATT + (size_t)h * SEQ * SEQ, XP + (size_t)h * HD,
                  nullptr, h ? MSG : nullptr, MSG,
                  SEQ, HD, SEQ,
                  SEQ, (long long)NHEAD * SEQ * SEQ,
                  NHEAD * HD, 1, (long long)SEQ * NHEAD * HD,
                  HD, (long long)SEQ * HD, 0, 1, NB,
                  1.0f, h ? 4 : 0);
    }
    gat_combine<<<(int)((Xn + 255) / 256), 256, 0, stream>>>(
        X, MSG, gat_b + (size_t)g * HD, (long long)Xn);
  }
  // emb = LN(enc + relu(x))
  layernorm<<<NB * SEQ, 256, 0, stream>>>(enc, X, gat_ln, gat_ln + HD, EMB, 1);

  // ---- transformer encoder ----
  for (int l = 0; l < LAYERS; ++l) {
    const float* Wl = attn_W + (size_t)l * 4 * HD * HD;
    const float* bl = attn_b + (size_t)l * 4 * HD;
    // Q/K/V projections, head-split output [B,NH,SEQ,Dh]
    launch_gemm(stream, EMB, Wl + 0 * HD * HD, bl + 0 * HD, nullptr, Q,
                NB * SEQ, HD, HD, HD, 0, HD, 1, 0, HD, 0, 0, 1, 1, 1.0f, 1 | 8);
    launch_gemm(stream, EMB, Wl + 1 * HD * HD, bl + 1 * HD, nullptr, Kt,
                NB * SEQ, HD, HD, HD, 0, HD, 1, 0, HD, 0, 0, 1, 1, 1.0f, 1 | 8);
    launch_gemm(stream, EMB, Wl + 2 * HD * HD, bl + 2 * HD, nullptr, V,
                NB * SEQ, HD, HD, HD, 0, HD, 1, 0, HD, 0, 0, 1, 1, 1.0f, 1 | 8);
    // scores = Q @ K^T / sqrt(Dh), batch = B*NH
    launch_gemm(stream, Q, Kt, nullptr, nullptr, ATT,
                SEQ, SEQ, HDH,
                HDH, (long long)SEQ * HDH,
                1, HDH, (long long)SEQ * HDH,
                SEQ, (long long)SEQ * SEQ, 0, 1, NB * NHEAD,
                0.17677669529663687f, 0);
    attn_softmax<<<NB * NHEAD * SEQ, 256, 0, stream>>>(ATT, adj);
    // ctx = attn @ V, heads merged back to [B,SEQ,H]
    launch_gemm(stream, ATT, V, nullptr, nullptr, CTX,
                SEQ, HDH, SEQ,
                SEQ, (long long)SEQ * SEQ,
                HDH, 1, (long long)SEQ * HDH,
                HD, (long long)SEQ * HD, HDH, NHEAD, NB * NHEAD, 1.0f, 0);
    // out-proj + residual, then LN (in place)
    launch_gemm(stream, CTX, Wl + 3 * HD * HD, bl + 3 * HD, EMB, EMB,
                NB * SEQ, HD, HD, HD, 0, HD, 1, 0, HD, 0, 0, 1, 1, 1.0f, 1 | 4);
    layernorm<<<NB * SEQ, 256, 0, stream>>>(
        EMB, nullptr, ln1 + (size_t)l * 2 * HD, ln1 + (size_t)l * 2 * HD + HD,
        EMB, 0);
    // FFN: relu(emb@W1+b1)@W2 + b2 + residual, then LN
    launch_gemm(stream, EMB, ffn_W1 + (size_t)l * HD * FFD, ffn_b1 + (size_t)l * FFD,
                nullptr, FF, NB * SEQ, FFD, HD,
                HD, 0, FFD, 1, 0, FFD, 0, 0, 1, 1, 1.0f, 1 | 2);
    launch_gemm(stream, FF, ffn_W2 + (size_t)l * FFD * HD, ffn_b2 + (size_t)l * HD,
                EMB, EMB, NB * SEQ, HD, FFD,
                FFD, 0, HD, 1, 0, HD, 0, 0, 1, 1, 1.0f, 1 | 4);
    layernorm<<<NB * SEQ, 256, 0, stream>>>(
        EMB, nullptr, ln2 + (size_t)l * 2 * HD, ln2 + (size_t)l * 2 * HD + HD,
        EMB, 0);
  }

  // ---- LSTM cell ----
  // gates = dec_in @ Wih^T  (B(k,n) = Wih[n*HD+k] -> ldbK=1, ldbN=HD)
  launch_gemm(stream, dec_in, lstm_Wih, nullptr, nullptr, GATES,
              NB, 4 * HD, HD, HD, 0, 1, HD, 0, 4 * HD, 0, 0, 1, 1, 1.0f, 0);
  launch_gemm(stream, dec_h, lstm_Whh, lstm_b, GATES, GATES,
              NB, 4 * HD, HD, HD, 0, 1, HD, 0, 4 * HD, 0, 0, 1, 1, 1.0f, 1 | 4);
  lstm_elem<<<(NB * HD + 255) / 256, 256, 0, stream>>>(GATES, dec_c, HO, CO);

  // ---- pointer head ----
  launch_gemm(stream, EMB, ptr_Wref, nullptr, nullptr, XP /*REF*/,
              NB * SEQ, HD, HD, HD, 0, HD, 1, 0, HD, 0, 0, 1, 1, 1.0f, 0);
  launch_gemm(stream, HO, ptr_Wq, nullptr, nullptr, QH,
              NB, HD, HD, HD, 0, HD, 1, 0, HD, 0, 0, 1, 1, 1.0f, 0);
  ptr_u<<<(NB * SEQ + 7) / 8, 256, 0, stream>>>(XP, QH, ptr_v, maskp, UU);
  policy_softmax<<<NB, 256, 0, stream>>>(UU, POL);
}
